// BiLSTMCRF_28552942584042
// MI455X (gfx1250) — matrix operand order repaired and checked by reading, not compiled, not run on hardware
//
#include <hip/hip_runtime.h>
#include <hip/hip_bf16.h>

// ---------------------------------------------------------------------------
// BiLSTM-CRF for MI455X (gfx1250), wave32 + WMMA bf16 + async global->LDS.
//   B=32  T=512  D=256  H=256  4H=1024  C=32
// ---------------------------------------------------------------------------

typedef __bf16 bf16_t;
typedef __attribute__((ext_vector_type(16))) __bf16 v16bf;
typedef __attribute__((ext_vector_type(8)))  float  v8f;

union Frag16 { v16bf v; uint4 q[2]; };
union Pack8  { uint4 q; bf16_t e[8]; };

#define T_ 512
#define B_ 32
#define D_ 256
#define H_ 256
#define G_ 1024   /* 4H */
#define C_ 32

__device__ __forceinline__ v8f wmma_bf16(v16bf a, v16bf b, v8f c) {
  return __builtin_amdgcn_wmma_f32_16x16x32_bf16(
      false, a, false, b, (short)0, c, false, false);
}

// Async copy of 16 bytes per lane: global -> LDS, tracked by ASYNCcnt.
// lds_off = LDS byte offset (low 32 bits of a flat shared pointer),
// gaddr   = 64-bit global address.
__device__ __forceinline__ void async_copy16(unsigned lds_off,
                                             const void* gaddr) {
  asm volatile("global_load_async_to_lds_b128 %0, %1, off"
               :: "v"(lds_off), "v"((unsigned long long)gaddr)
               : "memory");
}
__device__ __forceinline__ void async_wait0() {
  asm volatile("s_wait_asynccnt 0x0" ::: "memory");
}

// A fragment (16x32 bf16), row-major, leading dim ld (elements).
__device__ __forceinline__ v16bf load_a_frag(const bf16_t* base, int ld,
                                             int row0, int k0, int lane) {
  int r = lane & 15, half = lane >> 4;
  const bf16_t* p = base + (size_t)(row0 + r) * ld + k0 + half * 8;
  Frag16 f;
  f.q[0] = *(const uint4*)p;
  f.q[1] = *(const uint4*)(p + 16);
  return f.v;
}

// B fragment (32x16 bf16) from transposed weights WT[N][K] (row-major).
__device__ __forceinline__ v16bf load_b_frag(const bf16_t* base, int ld,
                                             int n0, int k0, int lane) {
  int r = lane & 15, half = lane >> 4;
  const bf16_t* p = base + (size_t)(n0 + r) * ld + k0 + half * 8;
  Frag16 f;
  f.q[0] = *(const uint4*)p;
  f.q[1] = *(const uint4*)(p + 16);
  return f.v;
}

__device__ __forceinline__ float sigf(float x) {
  return 1.0f / (1.0f + __expf(-x));
}
__device__ __forceinline__ float tanh_fast(float x) {
  x = fminf(15.0f, fmaxf(-15.0f, x));
  float e = __expf(2.0f * x);
  return (e - 1.0f) / (e + 1.0f);
}

// ---------------------------------------------------------------------------
__global__ void k_transpose_cast(const float* __restrict__ src,
                                 bf16_t* __restrict__ dst, int K, int N) {
  int idx = blockIdx.x * blockDim.x + threadIdx.x;
  if (idx >= N * K) return;
  int n = idx / K, k = idx - n * K;
  dst[idx] = (bf16_t)src[(size_t)k * N + n];
}

// ---------------------------------------------------------------------------
__global__ void k_embed(const int* __restrict__ chars,
                        const int* __restrict__ bigrams,
                        const float* __restrict__ ctab,
                        const float* __restrict__ btab,
                        bf16_t* __restrict__ X) {
  int idx = blockIdx.x * blockDim.x + threadIdx.x;  // T*B*256 total
  int d = idx & 255;
  int m = idx >> 8;           // t*B + b
  int t = m >> 5, b = m & 31;
  float v;
  if (d < 128) v = ctab[(size_t)chars[b * T_ + t] * 128 + d];
  else         v = btab[(size_t)bigrams[b * T_ + t] * 128 + (d - 128)];
  X[idx] = (bf16_t)v;
}

// ---------------------------------------------------------------------------
// gates_x = X[16384,256] @ WiT^T + bias, swizzled bf16 output.
// B tile (128 rows x 256 K = 64 KB) staged in LDS via async copy.
// grid = (16, 128), block = 256 (8 waves).
__global__ __launch_bounds__(256) void k_gemm_xwi(
    const bf16_t* __restrict__ X, const bf16_t* __restrict__ WiT,
    const float* __restrict__ bf_, const float* __restrict__ bb_,
    bf16_t* __restrict__ gx) {
  __shared__ bf16_t Bsh[128 * D_];   // 64 KB

  int tid = threadIdx.x;
  int lane = tid & 31;
  int w = tid >> 5;
  int gm = blockIdx.y * 8 + w;       // m-tile 0..1023
  int nt0 = blockIdx.x * 8;          // n-tile base 0..120
  int r = lane & 15;

  // async stage: rows nt0*16 .. nt0*16+127 of WiT (contiguous 64 KB)
  {
    unsigned lds_base = (unsigned)(unsigned long long)(uintptr_t)Bsh;
    const bf16_t* gsrc = WiT + (size_t)nt0 * 16 * D_;
#pragma unroll
    for (int it = 0; it < 16; ++it) {
      int chunk = it * 256 + tid;    // 16B chunks, 4096 total
      async_copy16(lds_base + chunk * 16, gsrc + chunk * 8);
    }
    async_wait0();
  }
  __syncthreads();

  v8f acc[8];
#pragma unroll
  for (int i = 0; i < 8; ++i) acc[i] = (v8f)0.0f;

  for (int k = 0; k < 8; ++k) {
    v16bf a = load_a_frag(X, D_, gm * 16, k * 32, lane);
#pragma unroll
    for (int i = 0; i < 8; ++i) {
      v16bf b = load_b_frag(Bsh, D_, i * 16, k * 32, lane);
      acc[i] = wmma_bf16(a, b, acc[i]);
    }
  }
#pragma unroll
  for (int i = 0; i < 8; ++i) {
    int n = (nt0 + i) * 16 + r;      // global gate column 0..2047
    float bias = (n < G_) ? bf_[n] : bb_[n - G_];
    Pack8 o;
#pragma unroll
    for (int p = 0; p < 8; ++p) o.e[p] = (bf16_t)(acc[i][p] + bias);
    *(uint4*)(gx + (((size_t)gm * 128 + (nt0 + i)) * 32 + lane) * 8) = o.q;
  }
}

// ---------------------------------------------------------------------------
// Persistent bidirectional LSTM. grid = 2 (dir), block = 512 (16 waves).
__global__ __launch_bounds__(512) void k_lstm(
    const bf16_t* __restrict__ gx, const bf16_t* __restrict__ WhT,
    const int* __restrict__ seq_len, bf16_t* __restrict__ feats) {
  __shared__ bf16_t hs[B_ * H_];     // 16 KB

  int dir = blockIdx.x;
  int tid = threadIdx.x, lane = tid & 31, w = tid >> 5;
  int r = lane & 15, half = lane >> 4;
  int mtile = w & 1, hblk = w >> 1;
  int m0 = mtile * 16;
  const bf16_t* Wd = WhT + (size_t)dir * G_ * H_;

  for (int i = tid; i < B_ * H_; i += 512) hs[i] = (bf16_t)0.0f;

  int sl[8];
#pragma unroll
  for (int p = 0; p < 8; ++p) sl[p] = seq_len[m0 + p + 8 * half];

  v8f c[2], h[2];
#pragma unroll
  for (int j = 0; j < 2; ++j) { c[j] = (v8f)0.0f; h[j] = (v8f)0.0f; }

  __syncthreads();

  for (int tt = 0; tt < T_; ++tt) {
    int t = dir ? (T_ - 1 - tt) : tt;

    // prefetch next timestep's gx tiles (latency-bound serial loop)
    {
      int tn = dir ? (t - 1) : (t + 1);
      if (tn >= 0 && tn < T_) {
#pragma unroll
        for (int gi = 0; gi < 4; ++gi)
#pragma unroll
          for (int j = 0; j < 2; ++j) {
            int ntg = dir * 64 + gi * 16 + hblk * 2 + j;
            size_t off =
                (((size_t)(tn * 2 + mtile) * 128 + ntg) * 32 + lane) * 8;
            __builtin_prefetch(gx + off, 0, 0);
          }
      }
    }

    v8f acc[4][2];
#pragma unroll
    for (int gi = 0; gi < 4; ++gi)
#pragma unroll
      for (int j = 0; j < 2; ++j) acc[gi][j] = (v8f)0.0f;

    // h @ Wh  (A from LDS, B streamed from L2-resident WhT)
    for (int k = 0; k < 8; ++k) {
      v16bf a = load_a_frag(hs, H_, m0, k * 32, lane);
#pragma unroll
      for (int gi = 0; gi < 4; ++gi)
#pragma unroll
        for (int j = 0; j < 2; ++j) {
          int ntl = gi * 16 + hblk * 2 + j;            // 0..63 within dir
          v16bf b = load_b_frag(Wd, H_, ntl * 16, k * 32, lane);
          acc[gi][j] = wmma_bf16(a, b, acc[gi][j]);
        }
    }

    // + precomputed (x@Wi + bias), swizzled: one b128 per tile
#pragma unroll
    for (int gi = 0; gi < 4; ++gi)
#pragma unroll
      for (int j = 0; j < 2; ++j) {
        int ntg = dir * 64 + gi * 16 + hblk * 2 + j;   // 0..127 global
        size_t off = (((size_t)(t * 2 + mtile) * 128 + ntg) * 32 + lane) * 8;
        Pack8 g; g.q = *(const uint4*)(gx + off);
#pragma unroll
        for (int p = 0; p < 8; ++p) acc[gi][j][p] += (float)g.e[p];
      }

    // gate fusion
#pragma unroll
    for (int j = 0; j < 2; ++j)
#pragma unroll
      for (int p = 0; p < 8; ++p) {
        float ig = sigf(acc[0][j][p]);
        float fg = sigf(acc[1][j][p]);
        float gg = tanh_fast(acc[2][j][p]);
        float og = sigf(acc[3][j][p]);
        float cn = fg * c[j][p] + ig * gg;
        float hn = og * tanh_fast(cn);
        bool mk = t < sl[p];
        c[j][p] = mk ? cn : c[j][p];
        h[j][p] = mk ? hn : h[j][p];
      }

    __syncthreads();  // all A-fragment reads of old h complete

#pragma unroll
    for (int j = 0; j < 2; ++j) {
      int col = hblk * 32 + j * 16 + r;
#pragma unroll
      for (int p = 0; p < 8; ++p) {
        int b = m0 + p + 8 * half;
        bf16_t hv = (bf16_t)h[j][p];
        hs[b * H_ + col] = hv;
        feats[((size_t)(t * B_ + b)) * (2 * H_) + dir * H_ + col] = hv;
      }
    }
    __syncthreads();  // new h visible before next step's reads
  }
}

// ---------------------------------------------------------------------------
// logits = log_softmax(feats @ out_W + out_b).  M=16384, K=512, N=32.
// out_W^T (32 KB) staged in LDS via async copy. grid = 128, block = 256.
__global__ __launch_bounds__(256) void k_proj(
    const bf16_t* __restrict__ feats, const bf16_t* __restrict__ oWT,
    const float* __restrict__ ob, float* __restrict__ logits) {
  __shared__ bf16_t osh[C_ * 2 * H_];  // 32 KB

  int tid = threadIdx.x;
  int lane = tid & 31, w = tid >> 5;
  int gm = blockIdx.x * 8 + w;       // m-tile 0..1023
  int r = lane & 15, half = lane >> 4;

  {
    unsigned lds_base = (unsigned)(unsigned long long)(uintptr_t)osh;
#pragma unroll
    for (int it = 0; it < 8; ++it) {
      int chunk = it * 256 + tid;    // 16B chunks, 2048 total
      async_copy16(lds_base + chunk * 16, oWT + chunk * 8);
    }
    async_wait0();
  }
  __syncthreads();

  v8f a0 = (v8f)0.0f, a1 = (v8f)0.0f;
  for (int k = 0; k < 16; ++k) {
    v16bf a = load_a_frag(feats, 2 * H_, gm * 16, k * 32, lane);
    v16bf b0 = load_b_frag(osh, 2 * H_, 0, k * 32, lane);
    v16bf b1 = load_b_frag(osh, 2 * H_, 16, k * 32, lane);
    a0 = wmma_bf16(a, b0, a0);
    a1 = wmma_bf16(a, b1, a1);
  }
  float bo0 = ob[r], bo1 = ob[16 + r];
#pragma unroll
  for (int p = 0; p < 8; ++p) {
    float v0 = a0[p] + bo0, v1 = a1[p] + bo1;
    float mx = fmaxf(v0, v1);
#pragma unroll
    for (int m = 8; m >= 1; m >>= 1) mx = fmaxf(mx, __shfl_xor(mx, m, 32));
    float s = __expf(v0 - mx) + __expf(v1 - mx);
#pragma unroll
    for (int m = 8; m >= 1; m >>= 1) s += __shfl_xor(s, m, 32);
    float ls = mx + __logf(s);
    int row = gm * 16 + p + 8 * half;
    logits[(size_t)row * C_ + r]      = v0 - ls;
    logits[(size_t)row * C_ + 16 + r] = v1 - ls;
  }
}

// ---------------------------------------------------------------------------
__global__ void k_gold(const int* __restrict__ target,
                       const int* __restrict__ seq_len,
                       const float* __restrict__ logits,
                       const float* __restrict__ trans,
                       const float* __restrict__ st,
                       const float* __restrict__ et,
                       float* __restrict__ gold) {
  int b = threadIdx.x;
  if (b >= B_) return;
  int sl = seq_len[b];
  float emit = 0.0f, trs = 0.0f;
  int prev = 0;
  for (int t = 0; t < T_; ++t) {
    if (t < sl) {
      int tg = target[b * T_ + t];
      emit += logits[((size_t)t * B_ + b) * C_ + tg];
      if (t >= 1) trs += trans[prev * C_ + tg];
      prev = tg;
    }
  }
  int first = target[b * T_];
  int last = target[b * T_ + (sl - 1)];
  gold[b] = emit + trs + st[first] + et[last];
}

// ---------------------------------------------------------------------------
__global__ __launch_bounds__(1024) void k_crf(
    const float* __restrict__ logits, const float* __restrict__ trans,
    const float* __restrict__ st, const float* __restrict__ et,
    const int* __restrict__ seq_len, const float* __restrict__ gold,
    float* __restrict__ out) {
  __shared__ float aL[B_ * C_];
  int tid = threadIdx.x, b = tid >> 5, j = tid & 31;

  float tc[C_];
#pragma unroll
  for (int i = 0; i < C_; ++i) tc[i] = trans[i * C_ + j];
  int sl = seq_len[b];

  aL[b * C_ + j] = logits[(size_t)b * C_ + j] + st[j];
  __syncthreads();

  for (int t = 1; t < T_; ++t) {
    float cur = aL[b * C_ + j];
    float na;
    if (t < sl) {
      float mx = -3.0e38f;
#pragma unroll
      for (int i = 0; i < C_; ++i) mx = fmaxf(mx, aL[b * C_ + i] + tc[i]);
      float s = 0.0f;
#pragma unroll
      for (int i = 0; i < C_; ++i) s += __expf(aL[b * C_ + i] + tc[i] - mx);
      na = logits[((size_t)t * B_ + b) * C_ + j] + mx + __logf(s);
    } else {
      na = cur;
    }
    __syncthreads();
    aL[b * C_ + j] = na;
    __syncthreads();
  }

  float v = aL[b * C_ + j] + et[j];
  float mx = v;
#pragma unroll
  for (int m = 16; m >= 1; m >>= 1) mx = fmaxf(mx, __shfl_xor(mx, m, 32));
  float s = __expf(v - mx);
#pragma unroll
  for (int m = 16; m >= 1; m >>= 1) s += __shfl_xor(s, m, 32);
  float norm = mx + __logf(s);
  if (j == 0) out[b] = norm - gold[b];
}

// ---------------------------------------------------------------------------
extern "C" void kernel_launch(void* const* d_in, const int* in_sizes, int n_in,
                              void* d_out, int out_size, void* d_ws,
                              size_t ws_size, hipStream_t stream) {
  (void)in_sizes; (void)n_in; (void)out_size; (void)ws_size;
  const int*   chars    = (const int*)d_in[0];
  const int*   bigrams  = (const int*)d_in[1];
  const int*   seq_len  = (const int*)d_in[2];
  const int*   target   = (const int*)d_in[3];
  const float* ctab     = (const float*)d_in[4];
  const float* btab     = (const float*)d_in[5];
  const float* Wi_f     = (const float*)d_in[6];
  const float* Wh_f     = (const float*)d_in[7];
  const float* b_f      = (const float*)d_in[8];
  const float* Wi_b     = (const float*)d_in[9];
  const float* Wh_b     = (const float*)d_in[10];
  const float* b_b      = (const float*)d_in[11];
  const float* out_W    = (const float*)d_in[12];
  const float* out_b    = (const float*)d_in[13];
  const float* trans    = (const float*)d_in[14];
  const float* st       = (const float*)d_in[15];
  const float* et       = (const float*)d_in[16];
  float* out = (float*)d_out;

  // workspace carve-up (~96.5 MB)
  char* ws = (char*)d_ws;
  bf16_t* Xbf    = (bf16_t*)(ws);                    // 8 MB
  bf16_t* WiT    = (bf16_t*)(ws + 8388608);          // 1 MB
  bf16_t* WhT    = (bf16_t*)(ws + 9437184);          // 1 MB
  bf16_t* oWT    = (bf16_t*)(ws + 10485760);         // 32 KB
  bf16_t* gx     = (bf16_t*)(ws + 10518528);         // 64 MB swizzled
  bf16_t* feats  = (bf16_t*)(ws + 77627392);         // 16 MB
  float*  logits = (float*)(ws + 94404608);          // 2 MB
  float*  gold   = (float*)(ws + 96501760);          // 128 B

  int nWi = G_ * D_;
  k_transpose_cast<<<(nWi + 255) / 256, 256, 0, stream>>>(Wi_f, WiT, D_, G_);
  k_transpose_cast<<<(nWi + 255) / 256, 256, 0, stream>>>(Wi_b, WiT + nWi, D_, G_);
  int nWh = G_ * H_;
  k_transpose_cast<<<(nWh + 255) / 256, 256, 0, stream>>>(Wh_f, WhT, H_, G_);
  k_transpose_cast<<<(nWh + 255) / 256, 256, 0, stream>>>(Wh_b, WhT + nWh, H_, G_);
  int nOW = C_ * 2 * H_;
  k_transpose_cast<<<(nOW + 255) / 256, 256, 0, stream>>>(out_W, oWT, 2 * H_, C_);

  k_embed<<<(T_ * B_ * D_) / 256, 256, 0, stream>>>(chars, bigrams, ctab, btab, Xbf);
  k_gemm_xwi<<<dim3(16, 128), 256, 0, stream>>>(Xbf, WiT, b_f, b_b, gx);
  k_lstm<<<2, 512, 0, stream>>>(gx, WhT, seq_len, feats);
  k_proj<<<128, 256, 0, stream>>>(feats, oWT, out_b, logits);
  k_gold<<<1, 32, 0, stream>>>(target, seq_len, logits, trans, st, et, gold);
  k_crf<<<1, 1024, 0, stream>>>(logits, trans, st, et, seq_len, gold, out);
}
